// Attention_17695265260026
// MI455X (gfx1250) — matrix operand order repaired and checked by reading, not compiled, not run on hardware
//
#include <hip/hip_runtime.h>
#include <hip/hip_bf16.h>

typedef _Float16 v16h __attribute__((ext_vector_type(16)));
typedef _Float16 v8h  __attribute__((ext_vector_type(8)));
typedef float    v8f  __attribute__((ext_vector_type(8)));

#define B_   2
#define S_   1024
#define D_   2048
#define H_   32
#define KV_  4
#define REP_ 8
#define HD_  64
#define SCALE_ 0.125f   // 1/sqrt(64)

// ---------------------------------------------------------------------------
// helpers
// ---------------------------------------------------------------------------
__device__ __forceinline__ v16h pack16(v8h lo, v8h hi) {
  v16h r;
#pragma unroll
  for (int i = 0; i < 8; ++i) { r[i] = lo[i]; r[i + 8] = hi[i]; }
  return r;
}

// ---------------------------------------------------------------------------
// elementwise conversions
// ---------------------------------------------------------------------------
__global__ void conv_f32_f16(const float* __restrict__ in, _Float16* __restrict__ out, int n) {
  int i = blockIdx.x * blockDim.x + threadIdx.x;
  if (i < n) out[i] = (_Float16)in[i];
}

// wT[n*K + k] = (f16) w[k*N + n]   (w is K x N row-major)
__global__ void transpose_f32_f16(const float* __restrict__ w, _Float16* __restrict__ wT,
                                  int K, int N) {
  int i = blockIdx.x * blockDim.x + threadIdx.x;
  if (i >= K * N) return;
  int k = i / N, n = i - k * N;
  wT[(size_t)n * K + k] = (_Float16)w[i];
}

// ---------------------------------------------------------------------------
// register-blocked wave-level WMMA GEMM:
//   C(f32, MxN) = A(f16, MxK) * Bt(f16, NxK)^T
// each wave computes a (16*MT) x (16*NT) block (MT x NT tiles, A reused NT
// times, B reused MT times); 4 waves per thread-block.
// ---------------------------------------------------------------------------
template <int MT, int NT>
__global__ void __launch_bounds__(128)
gemm_f16_wmma(const _Float16* __restrict__ A, const _Float16* __restrict__ Bt,
              float* __restrict__ C, int M, int N, int K) {
  const int wave = threadIdx.x >> 5;
  const int lane = threadIdx.x & 31;
  const int gw = blockIdx.x * 4 + wave;
  const int nb = N / (16 * NT), mb = M / (16 * MT);
  if (gw >= mb * nb) return;
  const int bm = gw / nb, bn = gw - bm * nb;
  const int l15 = lane & 15, hi = lane >> 4;

  const _Float16* pA[MT];
  const _Float16* pB[NT];
#pragma unroll
  for (int mt = 0; mt < MT; ++mt)
    pA[mt] = A + (size_t)(bm * 16 * MT + mt * 16 + l15) * K + hi * 8;   // A frag: klo = 8*hi
#pragma unroll
  for (int nt = 0; nt < NT; ++nt)
    pB[nt] = Bt + (size_t)(bn * 16 * NT + nt * 16 + l15) * K + hi * 8;  // B frag: klo = 8*hi

  v8f zero = {};
  v8f acc[MT][NT];
#pragma unroll
  for (int mt = 0; mt < MT; ++mt)
#pragma unroll
    for (int nt = 0; nt < NT; ++nt) acc[mt][nt] = zero;

#pragma unroll 1
  for (int k0 = 0; k0 < K; k0 += 32) {
    v16h aF[MT], bF[NT];
#pragma unroll
    for (int mt = 0; mt < MT; ++mt)
      aF[mt] = pack16(*(const v8h*)(pA[mt] + k0), *(const v8h*)(pA[mt] + k0 + 16));
#pragma unroll
    for (int nt = 0; nt < NT; ++nt)
      bF[nt] = pack16(*(const v8h*)(pB[nt] + k0), *(const v8h*)(pB[nt] + k0 + 16));
#pragma unroll
    for (int mt = 0; mt < MT; ++mt)
#pragma unroll
      for (int nt = 0; nt < NT; ++nt)
        acc[mt][nt] = __builtin_amdgcn_wmma_f32_16x16x32_f16(
            false, aF[mt], false, bF[nt], (short)0, acc[mt][nt], false, false);
  }

#pragma unroll
  for (int mt = 0; mt < MT; ++mt) {
    const int mrow = bm * 16 * MT + mt * 16 + 8 * hi;  // D layout: vgpr r -> row r + 8*(lane>>4)
#pragma unroll
    for (int nt = 0; nt < NT; ++nt) {
      const int col = bn * 16 * NT + nt * 16 + l15;
#pragma unroll
      for (int r = 0; r < 8; ++r)
        C[(size_t)(mrow + r) * N + col] = acc[mt][nt][r];
    }
  }
}

// ---------------------------------------------------------------------------
// RoPE + layout permute:  qraw f32 [B*S][H*64]  ->  qh f16 [B][H][S][64]
// ---------------------------------------------------------------------------
__global__ void rope_q(const float* __restrict__ qraw, const float* __restrict__ cosT,
                       const float* __restrict__ sinT, const int* __restrict__ posp,
                       _Float16* __restrict__ qh) {
  int idx = blockIdx.x * blockDim.x + threadIdx.x;
  if (idx >= B_ * S_ * H_ * 32) return;
  int d = idx & 31; int t = idx >> 5;
  int h = t % H_; t /= H_;
  int s = t % S_; int b = t / S_;
  int pos = posp[0];
  const float* row = qraw + (size_t)(b * S_ + s) * (H_ * HD_) + h * HD_;
  float t1 = row[d], t2 = row[d + 32];
  float c = cosT[(size_t)(pos + s) * 32 + d];
  float sn = sinT[(size_t)(pos + s) * 32 + d];
  _Float16* o = qh + (((size_t)b * H_ + h) * S_ + s) * HD_;
  o[d]      = (_Float16)(t1 * c - t2 * sn);
  o[d + 32] = (_Float16)(t2 * c + t1 * sn);
}

__global__ void rope_k(const float* __restrict__ kraw, const float* __restrict__ cosT,
                       const float* __restrict__ sinT, const int* __restrict__ posp,
                       _Float16* __restrict__ kh) {
  int idx = blockIdx.x * blockDim.x + threadIdx.x;
  if (idx >= B_ * S_ * KV_ * 32) return;
  int d = idx & 31; int t = idx >> 5;
  int kv = t % KV_; t /= KV_;
  int s = t % S_; int b = t / S_;
  int pos = posp[0];
  const float* row = kraw + (size_t)(b * S_ + s) * (KV_ * HD_) + kv * HD_;
  float t1 = row[d], t2 = row[d + 32];
  float c = cosT[(size_t)(pos + s) * 32 + d];
  float sn = sinT[(size_t)(pos + s) * 32 + d];
  _Float16* o = kh + (((size_t)b * KV_ + kv) * S_ + s) * HD_;
  o[d]      = (_Float16)(t1 * c - t2 * sn);
  o[d + 32] = (_Float16)(t2 * c + t1 * sn);
}

// vraw f32 [B*S][KV*64] -> vt f16 [B][KV][HD][S]  (pre-transposed for PV GEMM)
__global__ void conv_vt(const float* __restrict__ vraw, _Float16* __restrict__ vt) {
  int idx = blockIdx.x * blockDim.x + threadIdx.x;
  if (idx >= B_ * S_ * KV_ * HD_) return;
  int d = idx & 63; int t = idx >> 6;
  int kv = t % KV_; t /= KV_;
  int s = t % S_; int b = t / S_;
  vt[(((size_t)b * KV_ + kv) * HD_ + d) * S_ + s] =
      (_Float16)vraw[(size_t)(b * S_ + s) * (KV_ * HD_) + kv * HD_ + d];
}

// ---------------------------------------------------------------------------
// flash attention: one wave owns (b, h, 16-row query tile); 32-key tiles.
// qh [B][H][S][64], kh [B][KV][S][64], vt [B][KV][64][S] (f16)
//   -> ctx f16 [B*S][H*64]
// LDS used only for the P tile relayout (D layout -> A layout).
// ---------------------------------------------------------------------------
__global__ void __launch_bounds__(128)
attn_flash(const _Float16* __restrict__ qh, const _Float16* __restrict__ kh,
           const _Float16* __restrict__ vt, _Float16* __restrict__ ctx) {
  __shared__ __align__(16) _Float16 ldsP[4][16 * 32];  // per-wave P tile (1KB)
  const int wave = threadIdx.x >> 5;
  const int lane = threadIdx.x & 31;
  const int gw = blockIdx.x * 4 + wave;
  const int qtiles = S_ / 16;
  if (gw >= B_ * H_ * qtiles) return;
  const int qt = gw % qtiles;
  const int bh = gw / qtiles;
  const int h = bh % H_, b = bh / H_;
  const int kv = h / REP_;

  _Float16* P = &ldsP[wave][0];

  const _Float16* Q  = qh + (((size_t)b * H_  + h)  * S_) * HD_;
  const _Float16* Kp = kh + (((size_t)b * KV_ + kv) * S_) * HD_;
  const _Float16* Vt = vt + (((size_t)b * KV_ + kv) * HD_) * S_;

  const int l15 = lane & 15, hi = lane >> 4;

  // Q fragments for K-dim 0..31 and 32..63 (reused across all key tiles)
  v16h aq0, aq1;
  {
    const _Float16* pQ = Q + (size_t)(qt * 16 + l15) * HD_ + hi * 16;
    aq0 = pack16(*(const v8h*)(pQ),      *(const v8h*)(pQ + 8));
    aq1 = pack16(*(const v8h*)(pQ + 32), *(const v8h*)(pQ + 40));
  }

  float m_run[8], l_run[8];
  v8f o[4] = {};
#pragma unroll
  for (int r = 0; r < 8; ++r) { m_run[r] = -1e30f; l_run[r] = 0.f; }

  const int nkt = (qt >> 1) + 1;  // causal: key tiles of 32 needed for rows up to qt*16+15
  for (int kt = 0; kt < nkt; ++kt) {
    const int key0 = kt * 32;

    // scores: S = Q (16x64) x K^T (64 x 32), as two 16-col halves
    v8f s0 = {}, s1 = {};
    {
      const _Float16* pK0 = Kp + (size_t)(key0 + l15) * HD_ + hi * 16;       // keys  key0..+15
      const _Float16* pK1 = Kp + (size_t)(key0 + 16 + l15) * HD_ + hi * 16;  // keys +16..+31
      v16h b00 = pack16(*(const v8h*)(pK0),      *(const v8h*)(pK0 + 8));
      v16h b01 = pack16(*(const v8h*)(pK0 + 32), *(const v8h*)(pK0 + 40));
      v16h b10 = pack16(*(const v8h*)(pK1),      *(const v8h*)(pK1 + 8));
      v16h b11 = pack16(*(const v8h*)(pK1 + 32), *(const v8h*)(pK1 + 40));
      s0 = __builtin_amdgcn_wmma_f32_16x16x32_f16(false, aq0, false, b00, (short)0, s0, false, false);
      s0 = __builtin_amdgcn_wmma_f32_16x16x32_f16(false, aq1, false, b01, (short)0, s0, false, false);
      s1 = __builtin_amdgcn_wmma_f32_16x16x32_f16(false, aq0, false, b10, (short)0, s1, false, false);
      s1 = __builtin_amdgcn_wmma_f32_16x16x32_f16(false, aq1, false, b11, (short)0, s1, false, false);
    }

    // online softmax (row stats live replicated across each 16-lane group)
    const int keyA = key0 + l15, keyB = key0 + 16 + l15;
#pragma unroll
    for (int r = 0; r < 8; ++r) {
      const int qrow = qt * 16 + r + 8 * hi;
      float sa = s0[r] * SCALE_ + (keyA > qrow ? -1e9f : 0.f);
      float sb = s1[r] * SCALE_ + (keyB > qrow ? -1e9f : 0.f);
      float mx = fmaxf(sa, sb);
#pragma unroll
      for (int d = 1; d < 16; d <<= 1) mx = fmaxf(mx, __shfl_xor(mx, d, 32));
      const float mnew = fmaxf(m_run[r], mx);
      const float corr = __expf(m_run[r] - mnew);
      m_run[r] = mnew;
      const float pa = __expf(sa - mnew);
      const float pb = __expf(sb - mnew);
      float rs = pa + pb;
#pragma unroll
      for (int d = 1; d < 16; d <<= 1) rs += __shfl_xor(rs, d, 32);
      l_run[r] = l_run[r] * corr + rs;
      // store P tile (D layout position -> LDS row-major [16][32])
      P[(r + 8 * hi) * 32 + l15]      = (_Float16)pa;
      P[(r + 8 * hi) * 32 + 16 + l15] = (_Float16)pb;
      o[0][r] *= corr; o[1][r] *= corr; o[2][r] *= corr; o[3][r] *= corr;
    }

    // O += P (16x32) x V (32x64): A-frag of P from LDS, B-frags straight
    // from the pre-transposed vt (contiguous along keys).
    v16h ap;
    {
      const _Float16* pP = P + (size_t)l15 * 32 + hi * 16;
      ap = pack16(*(const v8h*)(pP), *(const v8h*)(pP + 8));
    }
#pragma unroll
    for (int nt = 0; nt < 4; ++nt) {
      const _Float16* pV = Vt + (size_t)(nt * 16 + l15) * S_ + key0 + hi * 16;
      v16h bv = pack16(*(const v8h*)(pV), *(const v8h*)(pV + 8));
      o[nt] = __builtin_amdgcn_wmma_f32_16x16x32_f16(false, ap, false, bv, (short)0, o[nt], false, false);
    }
  }

  // epilogue: normalize and write ctx[b*S + row][h*64 + col]
  const size_t orow = (size_t)b * S_ + qt * 16;
#pragma unroll
  for (int r = 0; r < 8; ++r) {
    const float inv = 1.0f / l_run[r];
#pragma unroll
    for (int nt = 0; nt < 4; ++nt) {
      ctx[(orow + r + 8 * hi) * (H_ * HD_) + h * HD_ + nt * 16 + l15] =
          (_Float16)(o[nt][r] * inv);
    }
  }
}

// ---------------------------------------------------------------------------
// launch
// ---------------------------------------------------------------------------
extern "C" void kernel_launch(void* const* d_in, const int* in_sizes, int n_in,
                              void* d_out, int out_size, void* d_ws, size_t ws_size,
                              hipStream_t stream) {
  const float* x    = (const float*)d_in[0];
  const float* wq   = (const float*)d_in[1];
  const float* wk   = (const float*)d_in[2];
  const float* wv   = (const float*)d_in[3];
  const float* wo   = (const float*)d_in[4];
  const float* cosT = (const float*)d_in[5];
  const float* sinT = (const float*)d_in[6];
  // d_in[7]: mask (unused — causal mask computed analytically)
  const int*   posp = (const int*)d_in[8];
  float* out = (float*)d_out;

  const int M  = B_ * S_;        // 2048 tokens
  const int NQ = H_ * HD_;       // 2048
  const int NK = KV_ * HD_;      // 256

  char* ws = (char*)d_ws;
  size_t off = 0;
  auto take = [&](size_t bytes) { void* p = ws + off; off += bytes; return p; };
  _Float16* xh   = (_Float16*)take((size_t)M * D_ * 2);     // 8 MiB
  _Float16* wqT  = (_Float16*)take((size_t)D_ * NQ * 2);    // 8 MiB
  _Float16* wkT  = (_Float16*)take((size_t)D_ * NK * 2);    // 1 MiB
  _Float16* wvT  = (_Float16*)take((size_t)D_ * NK * 2);    // 1 MiB
  _Float16* woT  = (_Float16*)take((size_t)NQ * D_ * 2);    // 8 MiB
  float*    qraw = (float*)take((size_t)M * NQ * 4);        // 16 MiB
  float*    kraw = (float*)take((size_t)M * NK * 4);        // 2 MiB
  float*    vraw = (float*)take((size_t)M * NK * 4);        // 2 MiB
  _Float16* qhd  = (_Float16*)take((size_t)M * NQ * 2);     // 8 MiB
  _Float16* khd  = (_Float16*)take((size_t)M * NK * 2);     // 1 MiB
  _Float16* vtd  = (_Float16*)take((size_t)M * NK * 2);     // 1 MiB
  _Float16* ctx  = (_Float16*)take((size_t)M * NQ * 2);     // 8 MiB  (total 64 MiB)

  const int T = 256;
  // convert / transpose
  conv_f32_f16<<<(M * D_ + T - 1) / T, T, 0, stream>>>(x, xh, M * D_);
  transpose_f32_f16<<<(D_ * NQ + T - 1) / T, T, 0, stream>>>(wq, wqT, D_, NQ);
  transpose_f32_f16<<<(D_ * NK + T - 1) / T, T, 0, stream>>>(wk, wkT, D_, NK);
  transpose_f32_f16<<<(D_ * NK + T - 1) / T, T, 0, stream>>>(wv, wvT, D_, NK);
  transpose_f32_f16<<<(NQ * D_ + T - 1) / T, T, 0, stream>>>(wo, woT, NQ, D_);

  // projections (register-blocked WMMA GEMMs, 64x64 per wave)
  {
    int waves = (M / 64) * (NQ / 64);
    gemm_f16_wmma<4, 4><<<(waves + 3) / 4, 128, 0, stream>>>(xh, wqT, qraw, M, NQ, D_);
  }
  {
    int waves = (M / 64) * (NK / 64);
    gemm_f16_wmma<4, 4><<<(waves + 3) / 4, 128, 0, stream>>>(xh, wkT, kraw, M, NK, D_);
    gemm_f16_wmma<4, 4><<<(waves + 3) / 4, 128, 0, stream>>>(xh, wvT, vraw, M, NK, D_);
  }

  // RoPE + head-major relayout (+ V pre-transpose)
  rope_q<<<(B_ * S_ * H_ * 32 + T - 1) / T, T, 0, stream>>>(qraw, cosT, sinT, posp, qhd);
  rope_k<<<(B_ * S_ * KV_ * 32 + T - 1) / T, T, 0, stream>>>(kraw, cosT, sinT, posp, khd);
  conv_vt<<<(B_ * S_ * KV_ * HD_ + T - 1) / T, T, 0, stream>>>(vraw, vtd);

  // flash attention (WMMA)
  {
    int waves = B_ * H_ * (S_ / 16);
    attn_flash<<<(waves + 3) / 4, 128, 0, stream>>>(qhd, khd, vtd, ctx);
  }

  // output projection -> d_out (f32)
  {
    int waves = (M / 64) * (D_ / 64);
    gemm_f16_wmma<4, 4><<<(waves + 3) / 4, 128, 0, stream>>>(ctx, woT, out, M, D_, NQ);
  }
  (void)in_sizes; (void)n_in; (void)out_size; (void)ws_size;
}
